// Sam3ViTRoPEAttention_73100343378041
// MI455X (gfx1250) — matrix operand order repaired and checked by reading, not compiled, not run on hardware
//
#include <hip/hip_runtime.h>
#include <hip/hip_bf16.h>

typedef __attribute__((ext_vector_type(16))) __bf16 v16bf;
typedef __attribute__((ext_vector_type(8)))  float  v8f;
typedef unsigned short u16;
typedef unsigned int   u32;

#define HID 1024
#define NH  16
#define HD  64
#define SEQ 1024
#define BATCH 4
#define ROWS (BATCH*SEQ)      // 4096
#define SCALE 0.125f          // 64^-0.5

union BFrag {
    v16bf v;
    uint4 q[2];
};

__device__ __forceinline__ u16 f2bf(float f) {
    u32 u = __float_as_uint(f);
    u32 r = u + 0x7FFFu + ((u >> 16) & 1u);   // round-to-nearest-even
    return (u16)(r >> 16);
}

// A fragment: 16x32 (MxK) bf16, row-major source, row stride ld (elements).
// Lanes 0-15 hold M=lane, K={0..7,16..23}; lanes 16-31 K={8..15,24..31}.
__device__ __forceinline__ v16bf load_A(const u16* __restrict__ base, int ld,
                                        int mBase, int kBase, int lane) {
    int m  = mBase + (lane & 15);
    int kh = (lane >> 4) << 3;               // 0 or 8
    const u16* p = base + (size_t)m * ld + kBase + kh;
    BFrag f;
    f.q[0] = *(const uint4*)(p);
    f.q[1] = *(const uint4*)(p + 16);
    return f.v;
}

// B fragment: 32x16 (KxN) where B[k][n] = W[n][k], W row-major [N][K] stride ld.
// Lane n = nBase+(lane&15); lanes 0-15 K={0..7,16..23}, lanes 16-31 K={8..15,24..31}.
__device__ __forceinline__ v16bf load_B(const u16* __restrict__ w, int ld,
                                        int nBase, int kBase, int lane) {
    int n  = nBase + (lane & 15);
    int ko = (lane >> 4) << 3;               // 0 or 8
    const u16* p = w + (size_t)n * ld + kBase + ko;
    BFrag f;
    f.q[0] = *(const uint4*)(p);
    f.q[1] = *(const uint4*)(p + 16);
    return f.v;
}

__global__ __launch_bounds__(256) void cvt_bf16_kernel(const float* __restrict__ src,
                                                       u16* __restrict__ dst, int n) {
    int i = blockIdx.x * blockDim.x + threadIdx.x;
    if (i < n) dst[i] = f2bf(src[i]);
}

// ---------------------------------------------------------------------------
// QKV projection: y = x @ w^T + b ; RoPE on q/k ; v stored transposed.
// grid (ROWS/64, HID/256, 3), block 256.
// Each wave computes a 64x32 output block (4 M-tiles x 2 N-tiles, 8 accums):
// 12 b128 loads feed 8 WMMAs per K-step; K-loop unrolled x2 so the second
// iteration's loads fill the WMMA->VALU hazard slots (no v_nop padding).
// ---------------------------------------------------------------------------
__global__ __launch_bounds__(256) void qkv_proj_kernel(
    const u16* __restrict__ x,
    const u16* __restrict__ wqb, const u16* __restrict__ wkb, const u16* __restrict__ wvb,
    const float* __restrict__ bq, const float* __restrict__ bk, const float* __restrict__ bv,
    const float* __restrict__ cosb, const float* __restrict__ sinb,
    u16* __restrict__ qout, u16* __restrict__ kout, u16* __restrict__ vTout)
{
    const int lane = threadIdx.x & 31;
    const int wave = threadIdx.x >> 5;
    const int mBase = blockIdx.x * 64;
    const int nBase = blockIdx.y * 256 + wave * 32;
    const int sel = blockIdx.z;
    const u16*   w    = (sel == 0) ? wqb : (sel == 1) ? wkb : wvb;
    const float* bias = (sel == 0) ? bq  : (sel == 1) ? bk  : bv;

    v8f acc[4][2];
#pragma unroll
    for (int i = 0; i < 4; ++i)
#pragma unroll
        for (int j = 0; j < 2; ++j) acc[i][j] = (v8f){};

#pragma unroll 2
    for (int kb = 0; kb < HID; kb += 32) {
        v16bf a[4], b[2];
#pragma unroll
        for (int i = 0; i < 4; ++i) a[i] = load_A(x, HID, mBase + 16 * i, kb, lane);
#pragma unroll
        for (int j = 0; j < 2; ++j) b[j] = load_B(w, HID, nBase + 16 * j, kb, lane);
#pragma unroll
        for (int i = 0; i < 4; ++i)
#pragma unroll
            for (int j = 0; j < 2; ++j)
                acc[i][j] = __builtin_amdgcn_wmma_f32_16x16x32_bf16(
                    false, a[i], false, b[j], (short)0, acc[i][j], false, false);
    }

    const int mOff = (lane >> 4) << 3;
#pragma unroll
    for (int j = 0; j < 2; ++j) {
        const int n = nBase + j * 16 + (lane & 15);
        const float bn = bias[n];
        const int h = n >> 6, d = n & 63;
        if (sel < 2) {
            u16* dst = (sel == 0) ? qout : kout;
#pragma unroll
            for (int i = 0; i < 4; ++i) {
#pragma unroll
                for (int r = 0; r < 8; ++r) {
                    int grow = mBase + i * 16 + mOff + r;   // b*SEQ + s
                    int s = grow & (SEQ - 1);
                    int bidx = grow >> 10;
                    float v0 = acc[i][j][r] + bn;
                    float partner = __shfl_xor(v0, 1, 32);       // feature n^1
                    float rot = (n & 1) ? partner : -partner;    // even:-x_odd, odd:+x_even
                    float c  = cosb[s * HD + d];
                    float sn = sinb[s * HD + d];
                    float o  = v0 * c + rot * sn;
                    size_t idx = ((size_t)((bidx * NH + h) * SEQ + s)) * HD + d;
                    dst[idx] = f2bf(o);
                }
            }
        } else {
#pragma unroll
            for (int i = 0; i < 4; ++i) {
#pragma unroll
                for (int r = 0; r < 8; ++r) {
                    int grow = mBase + i * 16 + mOff + r;
                    int s = grow & (SEQ - 1);
                    int bidx = grow >> 10;
                    float o = acc[i][j][r] + bn;
                    size_t idx = ((size_t)((bidx * NH + h) * HD + d)) * SEQ + s;
                    vTout[idx] = f2bf(o);
                }
            }
        }
    }
}

// ---------------------------------------------------------------------------
// Fused attention: scores -> softmax (LDS resident) -> P@V.
// grid (SEQ/16, BATCH*NH), block 256.
// ---------------------------------------------------------------------------
__global__ __launch_bounds__(256) void attn_kernel(
    const u16* __restrict__ q, const u16* __restrict__ k, const u16* __restrict__ vT,
    float* __restrict__ attn_w, u16* __restrict__ ao)
{
    __shared__ __align__(16) float S[16 * SEQ];      // 64 KB score/prob tile
    __shared__ __align__(16) u16   P[16 * SEQ];      // 32 KB bf16 probs
    __shared__ float red[256];                       // row reductions
    __shared__ float acc2[4 * 256];                  // PV K-split combine

    const int tid  = threadIdx.x;
    const int lane = tid & 31;
    const int wave = tid >> 5;
    const int bh    = blockIdx.y;
    const int qBase = blockIdx.x * 16;

    const u16* qh = q  + (size_t)bh * SEQ * HD;
    const u16* kh = k  + (size_t)bh * SEQ * HD;
    const u16* vh = vT + (size_t)bh * HD * SEQ;

    // ---- Phase 1: S = scale * q(16x64) @ k^T(64x1024) -------------------
    v16bf a0 = load_A(qh, HD, qBase, 0, lane);
    v16bf a1 = load_A(qh, HD, qBase, 32, lane);
#pragma unroll 2
    for (int i = 0; i < 8; ++i) {
        int nB = (wave * 8 + i) * 16;
        v8f acc = {};
        v16bf b0 = load_B(kh, HD, nB, 0, lane);
        acc = __builtin_amdgcn_wmma_f32_16x16x32_bf16(false, a0, false, b0,
                                                      (short)0, acc, false, false);
        v16bf b1 = load_B(kh, HD, nB, 32, lane);
        acc = __builtin_amdgcn_wmma_f32_16x16x32_bf16(false, a1, false, b1,
                                                      (short)0, acc, false, false);
        int n = nB + (lane & 15);
        int mOff = (lane >> 4) << 3;
#pragma unroll
        for (int r = 0; r < 8; ++r) S[(mOff + r) * SEQ + n] = acc[r] * SCALE;
    }
    __syncthreads();

    // ---- Phase 2: row softmax, write fp32 attn_w + bf16 P ---------------
    const int row = tid >> 4;                 // 0..15
    const int t   = tid & 15;
    float mx = -3.4e38f;
    for (int c = t; c < SEQ; c += 16) mx = fmaxf(mx, S[row * SEQ + c]);
    red[row * 16 + t] = mx;
    __syncthreads();
    float m = red[row * 16];
#pragma unroll
    for (int i = 1; i < 16; ++i) m = fmaxf(m, red[row * 16 + i]);
    __syncthreads();
    float sm = 0.f;
    for (int c = t; c < SEQ; c += 16) {
        float e = __expf(S[row * SEQ + c] - m);
        S[row * SEQ + c] = e;
        sm += e;
    }
    red[row * 16 + t] = sm;
    __syncthreads();
    float denom = 0.f;
#pragma unroll
    for (int i = 0; i < 16; ++i) denom += red[row * 16 + i];
    float inv = 1.f / denom;
    float* aw = attn_w + ((size_t)bh * SEQ + qBase + row) * SEQ;
    for (int c = t; c < SEQ; c += 16) {
        float p = S[row * SEQ + c] * inv;
        aw[c] = p;
        P[row * SEQ + c] = f2bf(p);
    }
    __syncthreads();

    // ---- Phase 3: out(16x64) = P(16x1024) @ V(1024x64), K split in 2 ----
    const int nt    = wave & 3;               // HD tile
    const int khalf = (wave >> 2) * 512;      // K half
    v8f acc = {};
#pragma unroll 2
    for (int ks = 0; ks < 16; ++ks) {
        int kb = khalf + ks * 32;
        v16bf a = load_A(P, SEQ, 0, kb, lane);               // from LDS
        v16bf b = load_B(vh, SEQ, nt * 16, kb, lane);        // vT rows are V columns
        acc = __builtin_amdgcn_wmma_f32_16x16x32_bf16(false, a, false, b,
                                                      (short)0, acc, false, false);
    }
    const int n = lane & 15;
    const int mOff = (lane >> 4) << 3;
    if (wave >= 4) {
#pragma unroll
        for (int r = 0; r < 8; ++r) acc2[nt * 256 + (mOff + r) * 16 + n] = acc[r];
    }
    __syncthreads();
    if (wave < 4) {
        int bidx = bh >> 4, h = bh & 15;
#pragma unroll
        for (int r = 0; r < 8; ++r) {
            float o = acc[r] + acc2[nt * 256 + (mOff + r) * 16 + n];
            int srow = qBase + mOff + r;
            size_t idx = ((size_t)(bidx * SEQ + srow)) * HID + (h * HD + nt * 16 + n);
            ao[idx] = f2bf(o);
        }
    }
}

// ---------------------------------------------------------------------------
// Output projection: out = ao @ wo^T + bo (fp32), 4Mx2N register tiling.
// grid (ROWS/64, HID/256), block 256.
// ---------------------------------------------------------------------------
__global__ __launch_bounds__(256) void out_proj_kernel(
    const u16* __restrict__ ao, const u16* __restrict__ wob,
    const float* __restrict__ bo, float* __restrict__ out)
{
    const int lane = threadIdx.x & 31;
    const int wave = threadIdx.x >> 5;
    const int mBase = blockIdx.x * 64;
    const int nBase = blockIdx.y * 256 + wave * 32;

    v8f acc[4][2];
#pragma unroll
    for (int i = 0; i < 4; ++i)
#pragma unroll
        for (int j = 0; j < 2; ++j) acc[i][j] = (v8f){};

#pragma unroll 2
    for (int kb = 0; kb < HID; kb += 32) {
        v16bf a[4], b[2];
#pragma unroll
        for (int i = 0; i < 4; ++i) a[i] = load_A(ao, HID, mBase + 16 * i, kb, lane);
#pragma unroll
        for (int j = 0; j < 2; ++j) b[j] = load_B(wob, HID, nBase + 16 * j, kb, lane);
#pragma unroll
        for (int i = 0; i < 4; ++i)
#pragma unroll
            for (int j = 0; j < 2; ++j)
                acc[i][j] = __builtin_amdgcn_wmma_f32_16x16x32_bf16(
                    false, a[i], false, b[j], (short)0, acc[i][j], false, false);
    }

    const int mOff = (lane >> 4) << 3;
#pragma unroll
    for (int j = 0; j < 2; ++j) {
        const int n = nBase + j * 16 + (lane & 15);
        const float bn = bo[n];
#pragma unroll
        for (int i = 0; i < 4; ++i) {
#pragma unroll
            for (int r = 0; r < 8; ++r) {
                int grow = mBase + i * 16 + mOff + r;
                out[(size_t)grow * HID + n] = acc[i][j][r] + bn;
            }
        }
    }
}

extern "C" void kernel_launch(void* const* d_in, const int* in_sizes, int n_in,
                              void* d_out, int out_size, void* d_ws, size_t ws_size,
                              hipStream_t stream) {
    const float* hs   = (const float*)d_in[0];
    const float* cosb = (const float*)d_in[1];
    const float* sinb = (const float*)d_in[2];
    const float* wq   = (const float*)d_in[3];
    const float* bq   = (const float*)d_in[4];
    const float* wk   = (const float*)d_in[5];
    const float* bk   = (const float*)d_in[6];
    const float* wv   = (const float*)d_in[7];
    const float* bv   = (const float*)d_in[8];
    const float* wo   = (const float*)d_in[9];
    const float* bo   = (const float*)d_in[10];

    char* ws = (char*)d_ws;
    const size_t XB = (size_t)ROWS * HID * 2;        // 8 MB
    const size_t WB = (size_t)HID * HID * 2;         // 2 MB
    u16* x_bf  = (u16*)(ws);
    u16* wq_bf = (u16*)(ws + XB);
    u16* wk_bf = (u16*)(ws + XB + WB);
    u16* wv_bf = (u16*)(ws + XB + 2 * WB);
    u16* wo_bf = (u16*)(ws + XB + 3 * WB);
    u16* qbuf  = (u16*)(ws + XB + 4 * WB);           // [B,NH,S,HD]
    u16* kbuf  = (u16*)(ws + 2 * XB + 4 * WB);
    u16* vTbuf = (u16*)(ws + 3 * XB + 4 * WB);       // [B,NH,HD,S]
    u16* aobuf = (u16*)(ws + 4 * XB + 4 * WB);       // [B*S, HID]

    float* out  = (float*)d_out;                     // (B,H,W,HID)
    float* attn = out + (size_t)ROWS * HID;          // (B,NH,S,S)

    // bf16 conversions
    cvt_bf16_kernel<<<(ROWS * HID + 255) / 256, 256, 0, stream>>>(hs, x_bf, ROWS * HID);
    cvt_bf16_kernel<<<(HID * HID + 255) / 256, 256, 0, stream>>>(wq, wq_bf, HID * HID);
    cvt_bf16_kernel<<<(HID * HID + 255) / 256, 256, 0, stream>>>(wk, wk_bf, HID * HID);
    cvt_bf16_kernel<<<(HID * HID + 255) / 256, 256, 0, stream>>>(wv, wv_bf, HID * HID);
    cvt_bf16_kernel<<<(HID * HID + 255) / 256, 256, 0, stream>>>(wo, wo_bf, HID * HID);

    // QKV projection + RoPE + V transpose
    qkv_proj_kernel<<<dim3(ROWS / 64, HID / 256, 3), 256, 0, stream>>>(
        x_bf, wq_bf, wk_bf, wv_bf, bq, bk, bv, cosb, sinb, qbuf, kbuf, vTbuf);

    // Fused attention (scores + softmax + PV), writes attn_w
    attn_kernel<<<dim3(SEQ / 16, BATCH * NH), 256, 0, stream>>>(
        qbuf, kbuf, vTbuf, attn, aobuf);

    // Output projection
    out_proj_kernel<<<dim3(ROWS / 64, HID / 256), 256, 0, stream>>>(
        aobuf, wo_bf, bo, out);
}
